// TOGL_66589172957709
// MI455X (gfx1250) — matrix operand order repaired and checked by reading, not compiled, not run on hardware
//
#include <hip/hip_runtime.h>

typedef __bf16 bf16_t;
typedef __attribute__((ext_vector_type(16))) __bf16 v16bf;
typedef __attribute__((ext_vector_type(8)))  __bf16 v8bf;
typedef __attribute__((ext_vector_type(8)))  float  v8f;
typedef __attribute__((ext_vector_type(4)))  float  v4f;

#define NN        1000000
#define NSEG      2000
#define BN_EPS    1e-5f
#define GRID_GEMM 2048    // grid for 256-thread GEMM kernels (grid-stride)
#define GRID_OUT  4096    // grid for 128-thread output kernel

// D = A(16x32 bf16) * B(32x16 bf16) + C(16x16 f32)
__device__ __forceinline__ v8f wmma_bf16(v16bf a, v16bf b, v8f c) {
  return __builtin_amdgcn_wmma_f32_16x16x32_bf16(false, a, false, b, (short)0, c,
                                                 false, false);
}

// A-operand fragment: ISA 16-bit A 16x32 layout. Lane holds row M=lane&15;
// halves j<8 -> K = k0 + 8*lhi + j ; j>=8 -> K = k0 + 16 + 8*lhi + (j-8).
// p points at element (row, k0 + 8*lhi); chunks at +0 and +16 elements.
__device__ __forceinline__ v16bf fragA(const bf16_t* p) {
  union { v16bf v; v8bf h[2]; } u;
  u.h[0] = *(const v8bf*)(p);
  u.h[1] = *(const v8bf*)(p + 16);
  return u.v;
}
// B-operand fragment: lane serves column n=lane&15; same K layout as fragA
// (j<8 -> K = k0+8*lhi+j; j>=8 -> K = k0+16+8*lhi+(j-8)). p points at &Bt[n][k0 + 8*lhi].
__device__ __forceinline__ v16bf fragB(const bf16_t* p) {
  union { v16bf v; v8bf h[2]; } u;
  u.h[0] = *(const v8bf*)(p);
  u.h[1] = *(const v8bf*)(p + 16);
  return u.v;
}

// ---------------- segment boundaries (batch is sorted) -------------------
__global__ __launch_bounds__(256) void k_bounds(const int* __restrict__ batch,
                                                int* __restrict__ segstart,
                                                int* __restrict__ segend) {
  const int i = blockIdx.x * 256 + threadIdx.x;
  if (i >= NN) return;
  const int b = batch[i];
  if (i == 0 || batch[i - 1] != b) segstart[b] = i;
  if (i == NN - 1 || batch[i + 1] != b) segend[b] = i + 1;
}

// -------- stage A: x0 = relu(pp_reshaped @ W0 + b0), bf16 out [N,64] ------
__global__ __launch_bounds__(256) void k_stage0(const float* __restrict__ pp,
                                                const float* __restrict__ W0,
                                                const float* __restrict__ b0,
                                                bf16_t* __restrict__ x0,
                                                int numTiles) {
  __shared__ alignas(16) bf16_t sB[64][40];      // Bt[n][k], K padded 16->32
  __shared__ float sBias[64];
  __shared__ alignas(16) bf16_t sOut[8][16][72]; // per-wave transpose staging
  const int tid = threadIdx.x;
  for (int i = tid; i < 64 * 40; i += 256) {
    int n = i / 40, k = i % 40;
    sB[n][k] = (k < 16) ? (bf16_t)W0[k * 64 + n] : (bf16_t)0.0f;
  }
  if (tid < 64) sBias[tid] = b0[tid];
  __syncthreads();
  const int lane = tid & 31, wave = tid >> 5;
  const int lcol = lane & 15, lhi = lane >> 4;
  // hoist loop-invariant B fragments + biases into registers
  v16bf bf[4]; float bias[4];
  #pragma unroll
  for (int ct = 0; ct < 4; ++ct) {
    bf[ct] = fragB(&sB[ct * 16 + lcol][8 * lhi]);
    bias[ct] = sBias[ct * 16 + lcol];
  }
  for (int tile = blockIdx.x; tile < numTiles; tile += gridDim.x) {
    const int base = tile * 128 + wave * 16;
    int node = base + lcol; if (node >= NN) node = NN - 1;
    v16bf a;
    #pragma unroll
    for (int f4 = 0; f4 < 4; ++f4) {   // features 8*lhi .. 8*lhi+7 = filt pairs
      const float2 pr = *(const float2*)(pp + ((size_t)(lhi * 4 + f4) * NN + node) * 2);
      a[2 * f4]     = (bf16_t)pr.x;
      a[2 * f4 + 1] = (bf16_t)pr.y;
    }
    #pragma unroll
    for (int j = 8; j < 16; ++j) a[j] = (bf16_t)0.0f;  // K=16..31 zero pad
    #pragma unroll
    for (int ct = 0; ct < 4; ++ct) {
      v8f acc = {};                       // inline-0 C operand
      acc = wmma_bf16(a, bf[ct], acc);
      #pragma unroll
      for (int r = 0; r < 8; ++r)
        sOut[wave][r + 8 * lhi][ct * 16 + lcol] = (bf16_t)fmaxf(acc[r] + bias[ct], 0.0f);
    }
    __syncthreads();
    #pragma unroll
    for (int pass = 0; pass < 4; ++pass) {   // coalesced b128 store-out
      const int r = pass * 4 + (lane >> 3), chunk = lane & 7;
      const int nd = base + r;
      if (nd < NN)
        *(v8bf*)(x0 + (size_t)nd * 64 + chunk * 8) = *(const v8bf*)&sOut[wave][r][chunk * 8];
    }
    __syncthreads();
  }
}

// ------- per-graph: xm = (segment_mean(h[N,64])) @ Wl[64,OUTD] -----------
template <int OUTD>
__global__ __launch_bounds__(256) void k_segproj(const bf16_t* __restrict__ h,
                                                 const int* __restrict__ segstart,
                                                 const int* __restrict__ segend,
                                                 const float* __restrict__ Wl,
                                                 float* __restrict__ xm) {
  __shared__ float partl[4][64];
  __shared__ float mean[64];
  const int g = blockIdx.x;
  const int s = segstart[g], e = segend[g];
  const int tid = threadIdx.x;
  const int col = tid & 63, pr = tid >> 6;
  float acc = 0.0f;
  for (int i = s + pr; i < e; i += 4) acc += (float)h[(size_t)i * 64 + col];
  partl[pr][col] = acc;
  __syncthreads();
  if (tid < 64) {
    const float cnt = (float)(e - s);
    const float m = partl[0][tid] + partl[1][tid] + partl[2][tid] + partl[3][tid];
    mean[tid] = (cnt > 0.5f) ? (m / cnt) : 0.0f;
  }
  __syncthreads();
  for (int o = tid; o < OUTD; o += 256) {
    float a = 0.0f;
    #pragma unroll 4
    for (int k = 0; k < 64; ++k) a += mean[k] * Wl[k * OUTD + o];
    xm[(size_t)g * OUTD + o] = a;
  }
}

// ---- layer1: h1 = relu(x0 @ g1_w + g1_b - xm1[batch]), bf16 [N,64] ------
__global__ __launch_bounds__(256) void k_layer1(const bf16_t* __restrict__ x0,
                                                const float* __restrict__ g1w,
                                                const float* __restrict__ g1b,
                                                const float* __restrict__ xm1,
                                                const int* __restrict__ batch,
                                                bf16_t* __restrict__ h1,
                                                int numTiles) {
  __shared__ alignas(16) bf16_t sB[64][72];
  __shared__ float sBias[64];
  __shared__ alignas(16) bf16_t sOut[8][16][72];
  const int tid = threadIdx.x;
  for (int i = tid; i < 64 * 72; i += 256) {
    int n = i / 72, k = i % 72;
    sB[n][k] = (k < 64) ? (bf16_t)g1w[k * 64 + n] : (bf16_t)0.0f;
  }
  if (tid < 64) sBias[tid] = g1b[tid];
  __syncthreads();
  const int lane = tid & 31, wave = tid >> 5;
  const int lcol = lane & 15, lhi = lane >> 4;
  // weights resident in registers: 8 fragments (64 VGPRs)
  v16bf bf0[4], bf1[4]; float bias[4];
  #pragma unroll
  for (int ct = 0; ct < 4; ++ct) {
    bf0[ct] = fragB(&sB[ct * 16 + lcol][8 * lhi]);
    bf1[ct] = fragB(&sB[ct * 16 + lcol][32 + 8 * lhi]);
    bias[ct] = sBias[ct * 16 + lcol];
  }
  for (int tile = blockIdx.x; tile < numTiles; tile += gridDim.x) {
    const int base = tile * 128 + wave * 16;
    int node = base + lcol; if (node >= NN) node = NN - 1;
    const bf16_t* arow = x0 + (size_t)node * 64 + 8 * lhi;
    v16bf a0 = fragA(arow);
    v16bf a1 = fragA(arow + 32);
    int bgs[8];
    #pragma unroll
    for (int r = 0; r < 8; ++r) {
      int nd = base + r + 8 * lhi; if (nd >= NN) nd = NN - 1;
      bgs[r] = batch[nd];
    }
    #pragma unroll
    for (int ct = 0; ct < 4; ++ct) {
      const int col = ct * 16 + lcol;
      v8f acc = {};
      acc = wmma_bf16(a0, bf0[ct], acc);
      acc = wmma_bf16(a1, bf1[ct], acc);
      #pragma unroll
      for (int r = 0; r < 8; ++r) {
        float v = fmaxf(acc[r] + bias[ct] - xm1[(size_t)bgs[r] * 64 + col], 0.0f);
        sOut[wave][r + 8 * lhi][col] = (bf16_t)v;
      }
    }
    __syncthreads();
    #pragma unroll
    for (int pass = 0; pass < 4; ++pass) {
      const int r = pass * 4 + (lane >> 3), chunk = lane & 7;
      const int nd = base + r;
      if (nd < NN)
        *(v8bf*)(h1 + (size_t)nd * 64 + chunk * 8) = *(const v8bf*)&sOut[wave][r][chunk * 8];
    }
    __syncthreads();
  }
}

// --- layer2 stats: v = relu(h1@g2_w + g2_b - xm2[batch]); per-block BN sums
__global__ __launch_bounds__(256) void k_layer2_stats(const bf16_t* __restrict__ h1,
                                                      const float* __restrict__ g2w,
                                                      const float* __restrict__ g2b,
                                                      const float* __restrict__ xm2,
                                                      const int* __restrict__ batch,
                                                      float* __restrict__ part,
                                                      int numTiles) {
  __shared__ alignas(16) bf16_t sB[128][72];
  __shared__ float sBias[128];
  __shared__ float sW[8][256];
  const int tid = threadIdx.x;
  for (int i = tid; i < 128 * 72; i += 256) {
    int n = i / 72, k = i % 72;
    sB[n][k] = (k < 64) ? (bf16_t)g2w[k * 128 + n] : (bf16_t)0.0f;
  }
  if (tid < 128) sBias[tid] = g2b[tid];
  __syncthreads();
  const int lane = tid & 31, wave = tid >> 5;
  const int lcol = lane & 15, lhi = lane >> 4;
  float lsum[8], lsq[8];
  #pragma unroll
  for (int ct = 0; ct < 8; ++ct) { lsum[ct] = 0.0f; lsq[ct] = 0.0f; }
  for (int tile = blockIdx.x; tile < numTiles; tile += gridDim.x) {
    const int base = tile * 128 + wave * 16;
    int node = base + lcol; if (node >= NN) node = NN - 1;
    const bf16_t* arow = h1 + (size_t)node * 64 + 8 * lhi;
    v16bf a0 = fragA(arow);
    v16bf a1 = fragA(arow + 32);
    int bgs[8]; bool valid[8];
    #pragma unroll
    for (int r = 0; r < 8; ++r) {
      int nd = base + r + 8 * lhi; valid[r] = (nd < NN); if (nd >= NN) nd = NN - 1;
      bgs[r] = batch[nd];
    }
    #pragma unroll
    for (int ct = 0; ct < 8; ++ct) {
      const int col = ct * 16 + lcol;
      v16bf b0f = fragB(&sB[col][8 * lhi]);
      v16bf b1f = fragB(&sB[col][32 + 8 * lhi]);
      v8f acc = {};
      acc = wmma_bf16(a0, b0f, acc);
      acc = wmma_bf16(a1, b1f, acc);
      const float bias = sBias[col];
      #pragma unroll
      for (int r = 0; r < 8; ++r) {
        float v = fmaxf(acc[r] + bias - xm2[(size_t)bgs[r] * 128 + col], 0.0f);
        if (valid[r]) { lsum[ct] += v; lsq[ct] += v * v; }
      }
    }
  }
  #pragma unroll
  for (int ct = 0; ct < 8; ++ct) {       // combine lane pairs (same column)
    lsum[ct] += __shfl_xor(lsum[ct], 16);
    lsq[ct]  += __shfl_xor(lsq[ct], 16);
  }
  if (lane < 16) {
    #pragma unroll
    for (int ct = 0; ct < 8; ++ct) {
      sW[wave][ct * 16 + lane]       = lsum[ct];
      sW[wave][128 + ct * 16 + lane] = lsq[ct];
    }
  }
  __syncthreads();
  float t = 0.0f;
  #pragma unroll
  for (int w = 0; w < 8; ++w) t += sW[w][tid];
  part[(size_t)blockIdx.x * 256 + tid] = t;   // [0..127]=sum, [128..255]=sumsq
}

// ---- reduce partials -> scale/shift for fused batch-norm -----------------
__global__ void k_finalize(const float* __restrict__ part,
                           const float* __restrict__ gamma,
                           const float* __restrict__ beta,
                           float* __restrict__ scaleshift) {
  const int c = threadIdx.x;  // 0..127
  float s = 0.0f, q = 0.0f;
  for (int p = 0; p < GRID_GEMM; ++p) {
    s += part[(size_t)p * 256 + c];
    q += part[(size_t)p * 256 + 128 + c];
  }
  const float invn = 1.0f / (float)NN;
  const float mean = s * invn;
  const float var = q * invn - mean * mean;
  const float sc = gamma[c] * rsqrtf(var + BN_EPS);
  scaleshift[c] = sc;
  scaleshift[128 + c] = beta[c] - mean * sc;
}

// ---- recompute v and emit out = x + v*scale + shift ----------------------
__global__ __launch_bounds__(128) void k_output(const bf16_t* __restrict__ h1,
                                                const float* __restrict__ g2w,
                                                const float* __restrict__ g2b,
                                                const float* __restrict__ xm2,
                                                const int* __restrict__ batch,
                                                const float* __restrict__ scaleshift,
                                                const float* __restrict__ x,
                                                float* __restrict__ out,
                                                int numTiles) {
  __shared__ alignas(16) bf16_t sB[128][72];
  __shared__ float sBias[128];
  __shared__ alignas(16) float sSc[128], sSh[128];
  __shared__ alignas(16) float sOut[4][16][132];
  const int tid = threadIdx.x;
  for (int i = tid; i < 128 * 72; i += 128) {
    int n = i / 72, k = i % 72;
    sB[n][k] = (k < 64) ? (bf16_t)g2w[k * 128 + n] : (bf16_t)0.0f;
  }
  if (tid < 128) {
    sBias[tid] = g2b[tid];
    sSc[tid] = scaleshift[tid];
    sSh[tid] = scaleshift[128 + tid];
  }
  __syncthreads();
  const int lane = tid & 31, wave = tid >> 5;
  const int lcol = lane & 15, lhi = lane >> 4;
  for (int tile = blockIdx.x; tile < numTiles; tile += gridDim.x) {
    const int base = tile * 64 + wave * 16;
    int node = base + lcol; if (node >= NN) node = NN - 1;
    const bf16_t* arow = h1 + (size_t)node * 64 + 8 * lhi;
    v16bf a0 = fragA(arow);
    v16bf a1 = fragA(arow + 32);
    int bgs[8];
    #pragma unroll
    for (int r = 0; r < 8; ++r) {
      int nd = base + r + 8 * lhi; if (nd >= NN) nd = NN - 1;
      bgs[r] = batch[nd];
    }
    #pragma unroll
    for (int ct = 0; ct < 8; ++ct) {
      const int col = ct * 16 + lcol;
      v16bf b0f = fragB(&sB[col][8 * lhi]);
      v16bf b1f = fragB(&sB[col][32 + 8 * lhi]);
      v8f acc = {};
      acc = wmma_bf16(a0, b0f, acc);
      acc = wmma_bf16(a1, b1f, acc);
      const float bias = sBias[col];
      #pragma unroll
      for (int r = 0; r < 8; ++r)
        sOut[wave][r + 8 * lhi][col] =
            fmaxf(acc[r] + bias - xm2[(size_t)bgs[r] * 128 + col], 0.0f);
    }
    __syncthreads();
    #pragma unroll
    for (int r = 0; r < 16; ++r) {   // coalesced: one 512B row per pass
      const int nd = base + r;
      if (nd < NN) {
        const int c4 = lane * 4;
        v4f v  = *(const v4f*)&sOut[wave][r][c4];
        v4f xv = *(const v4f*)(x + (size_t)nd * 128 + c4);
        v4f sc = *(const v4f*)&sSc[c4];
        v4f sh = *(const v4f*)&sSh[c4];
        v4f y;
        #pragma unroll
        for (int j = 0; j < 4; ++j) y[j] = xv[j] + v[j] * sc[j] + sh[j];
        *(v4f*)(out + (size_t)nd * 128 + c4) = y;
      }
    }
    __syncthreads();
  }
}

extern "C" void kernel_launch(void* const* d_in, const int* in_sizes, int n_in,
                              void* d_out, int out_size, void* d_ws, size_t ws_size,
                              hipStream_t stream) {
  (void)in_sizes; (void)n_in; (void)out_size; (void)ws_size;
  const float* x     = (const float*)d_in[0];
  const float* pp    = (const float*)d_in[1];
  const int*   batch = (const int*)d_in[2];
  const float* W0    = (const float*)d_in[3];
  const float* b0    = (const float*)d_in[4];
  const float* g1w   = (const float*)d_in[5];
  const float* g1b   = (const float*)d_in[6];
  const float* l1w   = (const float*)d_in[7];
  const float* g2w   = (const float*)d_in[8];
  const float* g2b   = (const float*)d_in[9];
  const float* l2w   = (const float*)d_in[10];
  const float* gamma = (const float*)d_in[11];
  const float* beta  = (const float*)d_in[12];
  float* out = (float*)d_out;
  char* ws = (char*)d_ws;

  // workspace layout (bytes), all 16B aligned
  float*  scaleshift = (float*)(ws);                 //      1024
  int*    segstart   = (int*)(ws + 1024);            //      8192
  int*    segend     = (int*)(ws + 9216);            //      8192
  float*  xm1        = (float*)(ws + 17408);         //   512,000
  float*  xm2        = (float*)(ws + 529408);        // 1,024,000
  float*  part       = (float*)(ws + 1553408);       // 2,097,152
  bf16_t* x0         = (bf16_t*)(ws + 3650560);      // 128,000,000
  bf16_t* h1         = (bf16_t*)(ws + 131650560);    // 128,000,000 (end ~259.7MB)

  hipMemsetAsync(ws + 1024, 0, 16384, stream);  // seg bounds (empty segs -> cnt 0)
  k_bounds<<<(NN + 255) / 256, 256, 0, stream>>>(batch, segstart, segend);

  const int nt128 = (NN + 127) / 128;  // 7813 tiles of 128 nodes
  k_stage0<<<GRID_GEMM, 256, 0, stream>>>(pp, W0, b0, x0, nt128);
  k_segproj<64><<<NSEG, 256, 0, stream>>>(x0, segstart, segend, l1w, xm1);
  k_layer1<<<GRID_GEMM, 256, 0, stream>>>(x0, g1w, g1b, xm1, batch, h1, nt128);
  k_segproj<128><<<NSEG, 256, 0, stream>>>(h1, segstart, segend, l2w, xm2);
  k_layer2_stats<<<GRID_GEMM, 256, 0, stream>>>(h1, g2w, g2b, xm2, batch, part, nt128);
  k_finalize<<<1, 128, 0, stream>>>(part, gamma, beta, scaleshift);
  const int nt64 = (NN + 63) / 64;     // 15625 tiles of 64 nodes
  k_output<<<GRID_OUT, 128, 0, stream>>>(h1, g2w, g2b, xm2, batch, scaleshift, x, out, nt64);
}
